// _DeformAlignWrapper_77945066488443
// MI455X (gfx1250) — matrix operand order repaired and checked by reading, compiled, not run on hardware
//
#include <hip/hip_runtime.h>

// ---------------- problem constants ----------------
#define BB    2
#define HH    128
#define WW    128
#define MIDC  64
#define DGRP  16
#define CIN   196          // 3*MID + 4
#define CPAD  224          // CIN padded to multiple of 32 (7 K-steps of 32)
#define COFF  432          // DG*27
#define HP    130          // H+2 (padded)
#define WP    130
#define KSTEPS 7           // CPAD/32
#define NT    27           // COFF/16 (real tiles)
#define NTP   28           // padded tiles -> 7 per N-group, no guard
#define DKK   1152         // 128*9 deform-GEMM K
#define DKS   36           // 1152/32
#define SLABW 34           // 32 pixels + 2 halo

// ---------------- workspace layout (bytes) ----------------
#define XP_OFF   0u
#define XP_BYTES (2u*HP*WP*CPAD*2u)                 // 15,142,400
#define WB_OFF   15142400u
#define WB_BYTES (28u*9u*7u*32u*16u*2u)             // 1,806,336
#define WD_OFF   16948736u
#define WD_BYTES (4u*36u*32u*16u*2u)                // 147,456
#define OF_OFF   17096192u
#define OF_BYTES (2u*288u*16384u*4u)                // 37,748,736
#define MK_OFF   54844928u

// B-tile stride between nt and nt+4 (in halves): 4 tiles * 9 kpos * 7 ks * 512
#define WB_TSTRIDE (4 * 9 * KSTEPS * 512)

typedef __attribute__((ext_vector_type(16))) _Float16 v16h;
typedef __attribute__((ext_vector_type(8)))  float    v8f;
typedef __attribute__((ext_vector_type(4)))  int      v4i;
union AV { v16h v; uint4 u[2]; };

#if __has_builtin(__builtin_amdgcn_global_load_async_to_lds_b128)
#define HAVE_ASYNC_LDS 1
typedef __attribute__((address_space(1))) v4i GV4;   // global int4 vector
typedef __attribute__((address_space(3))) v4i LV4;   // LDS int4 vector
#else
#define HAVE_ASYNC_LDS 0
#endif

static __device__ inline void wait_async_then_barrier() {
#if HAVE_ASYNC_LDS
#if __has_builtin(__builtin_amdgcn_s_wait_asynccnt)
    __builtin_amdgcn_s_wait_asynccnt(0);
#else
    asm volatile("s_wait_asynccnt 0x0" ::: "memory");
#endif
#endif
    __syncthreads();
}

// ================= prep: padded NHWC f16 image =================
__global__ __launch_bounds__(256) void prep_x_kernel(
    const float* __restrict__ ef, const float* __restrict__ f1,
    const float* __restrict__ f2, _Float16* __restrict__ Xp)
{
    int idx = blockIdx.x * 256 + threadIdx.x;
    const int total = 2 * HP * WP * CPAD;
    if (idx >= total) return;
    int c = idx % CPAD;
    int r = idx / CPAD;
    int x = r % WP; r /= WP;
    int y = r % HP; int b = r / HP;
    float val = 0.0f;
    if (c < CIN && y > 0 && y < HP - 1 && x > 0 && x < WP - 1) {
        int ys = y - 1, xs = x - 1;
        if (c < 192)      val = ef[((b * 192 + c) * HH + ys) * WW + xs];
        else if (c < 194) val = f1[((b * 2 + (c - 192)) * HH + ys) * WW + xs];
        else              val = f2[((b * 2 + (c - 194)) * HH + ys) * WW + xs];
    }
    Xp[idx] = (_Float16)val;
}

// ================= prep: conv weights -> B-tile per-lane layout =================
__global__ __launch_bounds__(256) void prep_wb_kernel(
    const float* __restrict__ w_off, _Float16* __restrict__ Wb)
{
    int idx = blockIdx.x * 256 + threadIdx.x;
    const int total = NTP * 9 * KSTEPS * 32 * 16;
    if (idx >= total) return;
    int j    = idx % 16;
    int lane = (idx / 16) % 32;
    int ks   = (idx / 512) % KSTEPS;
    int kpos = (idx / (512 * KSTEPS)) % 9;
    int nt   = idx / (512 * KSTEPS * 9);
    int n = nt * 16 + (lane & 15);
    int kk = ks * 32 + ((j < 8) ? ((lane < 16) ? j : 8 + j) : ((lane < 16) ? 16 + (j - 8) : 24 + (j - 8)));
    float val = (n < COFF && kk < CIN) ? w_off[n * (CIN * 9) + kk * 9 + kpos] : 0.0f;
    Wb[idx] = (_Float16)val;
}

// ================= prep: deform weights -> A-tile per-lane layout =================
__global__ __launch_bounds__(256) void prep_wd_kernel(
    const float* __restrict__ dcw, _Float16* __restrict__ Wd)
{
    int idx = blockIdx.x * 256 + threadIdx.x;
    const int total = 4 * DKS * 32 * 16;
    if (idx >= total) return;
    int j    = idx % 16;
    int lane = (idx / 16) % 32;
    int ks   = (idx / 512) % DKS;
    int mt   = idx / (512 * DKS);
    int o = mt * 16 + (lane & 15);
    int kl = (lane < 16) ? j : 16 + j;
    int i = ks * 32 + kl;
    Wd[idx] = (_Float16)dcw[o * DKK + i];
}

// ================= conv offset kernel (WMMA implicit GEMM) =================
// 256 threads = 8 waves; 32 pixels x 448(=432+pad) channels per block.
// wave w: M-tile = w>>2 (16 pixels), N-group = w&3 (7 of 28 tiles).
__global__ __launch_bounds__(256) void conv_offset_kernel(
    const _Float16* __restrict__ Xp, const _Float16* __restrict__ Wb,
    const float* __restrict__ b_off,
    const float* __restrict__ flow1, const float* __restrict__ flow2,
    float* __restrict__ OFF, float* __restrict__ MSK)
{
    __shared__ __align__(16) _Float16 slab[3 * SLABW * CPAD]; // 45,696 B

    int blk = blockIdx.x;
    int b  = blk >> 9;
    int rm = blk & 511;
    int y  = rm >> 2;
    int x0 = (rm & 3) << 5;

    int tid    = threadIdx.x;
    int wave   = __builtin_amdgcn_readfirstlane(tid) >> 5;
    int mtile  = wave >> 2;
    int ngroup = wave & 3;
    int lane   = tid & 31;

    // cooperative async load of 3 rows x 34 pixels x 224 ch
    const int CHUNKS = 3 * SLABW * (CPAD / 8); // 2856
    for (int q = tid; q < CHUNKS; q += 256) {
        int cio = q % (CPAD / 8);
        int p   = q / (CPAD / 8);
        int r   = p / SLABW;
        int col = p % SLABW;
        const uint4* src = (const uint4*)(Xp + (((b * HP) + y + r) * WP + (x0 + col)) * CPAD) + cio;
        uint4* dst = (uint4*)(slab + p * CPAD) + cio;
#if HAVE_ASYNC_LDS
        __builtin_amdgcn_global_load_async_to_lds_b128((GV4*)src, (LV4*)dst, 0, 0);
#else
        *dst = *src;
#endif
    }
    wait_async_then_barrier();

    v8f acc[7] = {};

    auto loadB = [&](AV* bm, int step) {
        int kpos = step / 7, ks = step % 7;
        const _Float16* bbase =
            Wb + (((ngroup * 9 + kpos) * KSTEPS + ks) * 32 + lane) * 16;
        #pragma unroll
        for (int t = 0; t < 7; ++t) {
            bm[t].u[0] = *(const uint4*)(bbase + t * WB_TSTRIDE);
            bm[t].u[1] = *(const uint4*)(bbase + t * WB_TSTRIDE + 8);
        }
    };
    auto loadA = [&](AV& a, int step) {
        int kpos = step / 7, ks = step % 7;
        int ky = kpos / 3, kx = kpos % 3;
        int m  = (lane & 15) + (mtile << 4);
        int c0 = ks * 32 + ((lane < 16) ? 0 : 8);
        const _Float16* ap = &slab[((ky * SLABW) + kx + m) * CPAD + c0];
        a.u[0] = *(const uint4*)ap;
        a.u[1] = *(const uint4*)(ap + 16);
    };

    // two-step software pipeline over 63 K-steps
    AV b0[7], b1[7], a0, a1;
    loadB(b0, 0); loadA(a0, 0);
    for (int s = 0; s < 63; s += 2) {
        if (s + 1 < 63) { loadB(b1, s + 1); loadA(a1, s + 1); }
        #pragma unroll
        for (int t = 0; t < 7; ++t)
            acc[t] = __builtin_amdgcn_wmma_f32_16x16x32_f16(
                false, a0.v, false, b0[t].v, (short)0, acc[t], false, false);
        if (s + 1 < 63) {
            if (s + 2 < 63) { loadB(b0, s + 2); loadA(a0, s + 2); }
            #pragma unroll
            for (int t = 0; t < 7; ++t)
                acc[t] = __builtin_amdgcn_wmma_f32_16x16x32_f16(
                    false, a1.v, false, b1[t].v, (short)0, acc[t], false, false);
        }
    }

    // fused epilogue: bias + 10*tanh + flipped-flow add, or sigmoid for mask
    for (int t = 0; t < 7; ++t) {
        int nt = ngroup + 4 * t;
        if (nt >= NT) continue;                 // only the zero-pad tile skipped
        int n = nt * 16 + (lane & 15);
        float bias = b_off[n];
        #pragma unroll
        for (int v = 0; v < 8; ++v) {
            int mrow = v + ((lane < 16) ? 0 : 8);
            int x = x0 + (mtile << 4) + mrow;
            float val = acc[t][v] + bias;
            if (n < 288) {
                const float* fl = (n < 144) ? flow1 : flow2;
                int d = (n & 1) ? 0 : 1;        // even offset ch -> dy = flow[:,1]
                float f = fl[((b * 2 + d) * HH + y) * WW + x];
                OFF[((b * 288 + n) * HH + y) * WW + x] = 10.0f * tanhf(val) + f;
            } else {
                MSK[((b * 144 + (n - 288)) * HH + y) * WW + x] =
                    1.0f / (1.0f + __expf(-val));
            }
        }
    }
}

// ================= deformable conv kernel =================
__global__ __launch_bounds__(256) void deform_kernel(
    const float* __restrict__ OFF, const float* __restrict__ MSK,
    const float* __restrict__ fpv, const float* __restrict__ fn2,
    const _Float16* __restrict__ Wd, const float* __restrict__ dc_bias,
    float* __restrict__ out)
{
    __shared__ __align__(16) _Float16 sm[16 * DKK]; // 36,864 B

    int blk = blockIdx.x;
    int b  = blk >> 10;
    int rm = blk & 1023;
    int y  = rm >> 3;
    int x0 = (rm & 7) << 4;

    int tid  = threadIdx.x;
    int wave = __builtin_amdgcn_readfirstlane(tid) >> 5;
    int lane = tid & 31;

    // ---- phase 1: 2304 tasks = (16 pixels) x (16 groups) x (9 taps) ----
    for (int task = tid; task < 16 * DGRP * 9; task += 256) {
        int p  = task / (DGRP * 9);
        int gk = task % (DGRP * 9);
        int g  = gk / 9;
        int k  = gk % 9;
        int ky = k / 3, kx = k % 3;
        int x  = x0 + p;

        int och = (g * 9 + k) * 2;
        float dy = OFF[((b * 288 + och) * HH + y) * WW + x];
        float dx = OFF[((b * 288 + och + 1) * HH + y) * WW + x];
        float mk = MSK[((b * 144 + g * 9 + k) * HH + y) * WW + x];

        float py = (float)(y + ky - 1) + dy;
        float px = (float)(x + kx - 1) + dx;
        float fy0 = floorf(py), fx0 = floorf(px);
        int iy0 = (int)fy0, ix0 = (int)fx0;
        float ay = py - fy0, ax = px - fx0;
        int iy1 = iy0 + 1, ix1 = ix0 + 1;
        float vy0 = (iy0 >= 0 && iy0 < HH) ? 1.0f : 0.0f;
        float vy1 = (iy1 >= 0 && iy1 < HH) ? 1.0f : 0.0f;
        float vx0 = (ix0 >= 0 && ix0 < WW) ? 1.0f : 0.0f;
        float vx1 = (ix1 >= 0 && ix1 < WW) ? 1.0f : 0.0f;
        int cy0 = iy0 < 0 ? 0 : (iy0 > HH - 1 ? HH - 1 : iy0);
        int cy1 = iy1 < 0 ? 0 : (iy1 > HH - 1 ? HH - 1 : iy1);
        int cx0 = ix0 < 0 ? 0 : (ix0 > WW - 1 ? WW - 1 : ix0);
        int cx1 = ix1 < 0 ? 0 : (ix1 > WW - 1 ? WW - 1 : ix1);
        float w00 = (1.0f - ay) * (1.0f - ax) * vy0 * vx0;
        float w01 = (1.0f - ay) * ax * vy0 * vx1;
        float w10 = ay * (1.0f - ax) * vy1 * vx0;
        float w11 = ay * ax * vy1 * vx1;

        #pragma unroll
        for (int c = 0; c < 8; ++c) {
            int cin = g * 8 + c;
            const float* base = (cin < MIDC)
                ? (fpv + (b * MIDC + cin) * (HH * WW))
                : (fn2 + (b * MIDC + (cin - MIDC)) * (HH * WW));
            float v = w00 * base[cy0 * WW + cx0] + w01 * base[cy0 * WW + cx1]
                    + w10 * base[cy1 * WW + cx0] + w11 * base[cy1 * WW + cx1];
            sm[p * DKK + cin * 9 + k] = (_Float16)(v * mk);
        }
    }
    __syncthreads();

    // ---- phase 2: GEMM out[64][16] = Wd[64][1152] * sm[1152][16] ----
    if (wave < 4) {
        int mt = wave;
        auto loadAB = [&](AV& a, AV& bm, int ks) {
            const _Float16* ap = Wd + ((mt * DKS + ks) * 32 + lane) * 16;
            a.u[0] = *(const uint4*)ap;
            a.u[1] = *(const uint4*)(ap + 8);
            int i0 = ks * 32 + ((lane < 16) ? 0 : 16);
            const _Float16* bp = &sm[(lane & 15) * DKK + i0];
            bm.u[0] = *(const uint4*)bp;
            bm.u[1] = *(const uint4*)(bp + 8);
        };
        v8f acc = {};
        AV a0, a1, b0, b1;
        loadAB(a0, b0, 0);
        for (int ks = 0; ks < DKS; ks += 2) {   // DKS even -> clean pipeline
            loadAB(a1, b1, ks + 1);
            acc = __builtin_amdgcn_wmma_f32_16x16x32_f16(
                false, a0.v, false, b0.v, (short)0, acc, false, false);
            if (ks + 2 < DKS) loadAB(a0, b0, ks + 2);
            acc = __builtin_amdgcn_wmma_f32_16x16x32_f16(
                false, a1.v, false, b1.v, (short)0, acc, false, false);
        }
        int p = lane & 15;
        int x = x0 + p;
        #pragma unroll
        for (int v = 0; v < 8; ++v) {
            int o = mt * 16 + v + ((lane < 16) ? 0 : 8);
            out[((b * MIDC + o) * HH + y) * WW + x] = acc[v] + dc_bias[o];
        }
    }
}

// ================= host launcher =================
extern "C" void kernel_launch(void* const* d_in, const int* in_sizes, int n_in,
                              void* d_out, int out_size, void* d_ws, size_t ws_size,
                              hipStream_t stream) {
    const float* extra_feat = (const float*)d_in[0];
    const float* flow_1     = (const float*)d_in[1];
    const float* flow_2     = (const float*)d_in[2];
    const float* feat_prop  = (const float*)d_in[3];
    const float* feat_n2    = (const float*)d_in[4];
    const float* w_off      = (const float*)d_in[5];
    const float* b_off      = (const float*)d_in[6];
    const float* dc_weight  = (const float*)d_in[7];
    const float* dc_bias    = (const float*)d_in[8];
    float* out = (float*)d_out;

    char* ws = (char*)d_ws;
    _Float16* Xp = (_Float16*)(ws + XP_OFF);
    _Float16* Wb = (_Float16*)(ws + WB_OFF);
    _Float16* Wd = (_Float16*)(ws + WD_OFF);
    float*    OF = (float*)(ws + OF_OFF);
    float*    MK = (float*)(ws + MK_OFF);

    {
        int total = 2 * HP * WP * CPAD;
        prep_x_kernel<<<(total + 255) / 256, 256, 0, stream>>>(extra_feat, flow_1, flow_2, Xp);
    }
    {
        int total = NTP * 9 * KSTEPS * 32 * 16;
        prep_wb_kernel<<<(total + 255) / 256, 256, 0, stream>>>(w_off, Wb);
    }
    {
        int total = 4 * DKS * 32 * 16;
        prep_wd_kernel<<<(total + 255) / 256, 256, 0, stream>>>(dc_weight, Wd);
    }

    conv_offset_kernel<<<1024, 256, 0, stream>>>(Xp, Wb, b_off, flow_1, flow_2, OF, MK);
    deform_kernel<<<2048, 256, 0, stream>>>(OF, MK, feat_prop, feat_n2, Wd, dc_bias, out);
}
